// GeneratorRNN_16398185136329
// MI455X (gfx1250) — matrix operand order repaired and checked by reading, not compiled, not run on hardware
//
#include <hip/hip_runtime.h>

// ---------------------------------------------------------------------------
// GeneratorRNN (LSTM decoder) for gfx1250 / MI455X.
//   B=65536, LATENT=128, HID=256, SEQ=50, NUC=4, SOS=4
// bf16 WMMA 16x16x32 recurrent GEMM. c-state in WMMA C fragments (registers),
// h/x staged in LDS (bf16, ping-pong, conflict-free padded strides).
// Weight B-tiles staged cooperatively into LDS (shared by all 8 waves) with a
// VGPR prefetch pipeline + split s_barrier_signal/wait, cutting L2 weight
// traffic 8x vs per-wave streaming.
// ---------------------------------------------------------------------------

typedef __bf16         v16bf __attribute__((ext_vector_type(16)));
typedef float          v8f   __attribute__((ext_vector_type(8)));
typedef float          f32x4 __attribute__((ext_vector_type(4)));
typedef unsigned int   u32x4 __attribute__((ext_vector_type(4)));

union Frag16 { v16bf v; u32x4 u[2]; };

#define BATCH    65536
#define LATENT   128
#define HID      256
#define GATES    1024          // 4*HID
#define KTOT     512           // HID (x) + HID (h)
#define SEQ      50
#define NUC      4
#define SOS      4

#define WAVES    8
#define ROWS_PW  16            // batch rows per wave (one WMMA M tile)
#define ROWS_WG  128           // rows per workgroup
#define ASTRIDE  776           // A row stride (bf16): x[0,256) hA[256,512) hB[512,768) pad 8
                               // 776*2=1552 B, 1552%256=16 -> conflict-free b128 frags
#define BSTRIDE  520           // B column stride (bf16): 512 K + pad 8
                               // 520*2=1040 B, 1040%256=16 -> conflict-free b128 frags

// d_ws layout (bytes)
#define OFF_WT    0u           // Wt[1024][512] bf16 (transposed; k<256 = W_ih, k>=256 = W_hh)
#define OFF_BG    1048576u     // b_ih+b_hh combined, f32[1024]
#define OFF_EMB   1052672u     // emb bf16[5*256]
#define OFF_WLHT  1056768u     // W_lh^T f32[256][128]
#define OFF_WLCT  1187840u     // W_lc^T f32[256][128]

#define LDS_A_BYTES (WAVES * ROWS_PW * ASTRIDE * 2)   // 198656
#define LDS_B_BYTES (64 * BSTRIDE * 2)                // 66560
#define SMEM_BYTES  (LDS_A_BYTES + LDS_B_BYTES)       // 265216 < 320KB

__device__ __forceinline__ unsigned short f2bf(float f) {
    unsigned u = __float_as_uint(f);
    unsigned r = u + 0x7FFFu + ((u >> 16) & 1u);   // round-to-nearest-even
    return (unsigned short)(r >> 16);
}
__device__ __forceinline__ float bf2f(unsigned short h) {
    return __uint_as_float(((unsigned)h) << 16);
}
__device__ __forceinline__ float sigm(float x) { return 1.0f / (1.0f + __expf(-x)); }
__device__ __forceinline__ unsigned hashu(unsigned a) {
    a ^= a >> 16; a *= 0x7feb352du;
    a ^= a >> 15; a *= 0x846ca68bu;
    a ^= a >> 16; return a;
}

// Workgroup split barrier without the conservative loadcnt flush of
// __syncthreads(): in-flight global prefetch loads keep running across it.
__device__ __forceinline__ void wg_barrier() {
    asm volatile("s_barrier_signal -1\n\ts_barrier_wait -1" ::: "memory");
}
// Release own LDS stores before signalling peers.
__device__ __forceinline__ void lds_release_barrier() {
    asm volatile("s_wait_dscnt 0x0\n\ts_barrier_signal -1\n\ts_barrier_wait -1" ::: "memory");
}

// ---------------------------------------------------------------------------
// One-shot weight prep: transpose+convert recurrent weights to bf16, combine
// biases, convert embedding, transpose latent projections (f32).
// ---------------------------------------------------------------------------
__global__ __launch_bounds__(256) void prep_kernel(
    const float* __restrict__ W_lh, const float* __restrict__ W_lc,
    const float* __restrict__ emb,  const float* __restrict__ W_ih,
    const float* __restrict__ W_hh, const float* __restrict__ b_ih,
    const float* __restrict__ b_hh, char* __restrict__ ws)
{
    unsigned short* Wt   = (unsigned short*)(ws + OFF_WT);
    float*          bg   = (float*)(ws + OFF_BG);
    unsigned short* ebf  = (unsigned short*)(ws + OFF_EMB);
    float*          Wlht = (float*)(ws + OFF_WLHT);
    float*          Wlct = (float*)(ws + OFF_WLCT);

    int idx = blockIdx.x * 256 + threadIdx.x;        // 0 .. 1024*512-1
    int n = idx >> 9;        // gate column 0..1023
    int k = idx & 511;       // k 0..511
    float w = (k < HID) ? W_ih[(size_t)k * GATES + n]
                        : W_hh[(size_t)(k - HID) * GATES + n];
    Wt[idx] = f2bf(w);

    if (idx < GATES)         bg[idx] = b_ih[idx] + b_hh[idx];
    if (idx < 5 * HID)       ebf[idx] = f2bf(emb[idx]);
    if (idx < HID * LATENT) {                        // 32768
        int col = idx >> 7, kk = idx & 127;          // col 0..255, kk 0..127
        Wlht[idx] = W_lh[(size_t)kk * HID + col];
        Wlct[idx] = W_lc[(size_t)kk * HID + col];
    }
}

// ---------------------------------------------------------------------------
// Main persistent LSTM kernel: 512 blocks x 256 threads (8 waves, 128 rows/WG)
// ---------------------------------------------------------------------------
__global__ __launch_bounds__(256) void rnn_kernel(
    const float* __restrict__ z,     const float* __restrict__ b_lh,
    const float* __restrict__ b_lc,  const float* __restrict__ W_out,
    const float* __restrict__ b_out, const unsigned short* __restrict__ Wt,
    const float* __restrict__ bg,    const unsigned short* __restrict__ ebf,
    const float* __restrict__ Wlht,  const float* __restrict__ Wlct,
    float* __restrict__ out)
{
    extern __shared__ char smem[];
    unsigned short* a_all = (unsigned short*)smem;                  // A: x/h per wave
    unsigned short* ldsB  = (unsigned short*)(smem + LDS_A_BYTES);  // B: 64 cols x 520

    const int tid  = threadIdx.x;
    const int wave = tid >> 5;
    const int lane = tid & 31;
    const int half = lane >> 4;      // 0: lanes 0-15, 1: lanes 16-31
    const int l16  = lane & 15;
    unsigned short* aw = a_all + wave * ROWS_PW * ASTRIDE;
    const int grow0 = blockIdx.x * ROWS_WG + wave * ROWS_PW;

    // staging geometry: thread covers 16B chunk kc of columns (j*4 + c0)
    const int kc = tid & 63;         // 16B chunk within a column (k = kc*8)
    const int c0 = tid >> 6;         // column offset 0..3

    // ---- x = emb[SOS] for all 16 rows of this wave ----
    {
        const u32x4* ep = (const u32x4*)(ebf + SOS * HID + half * 128);
        u32x4* xp = (u32x4*)(aw + l16 * ASTRIDE + half * 128);
        #pragma unroll
        for (int j = 0; j < 16; ++j) xp[j] = ep[j];
    }

    // ---- h0 = tanh(z@W_lh+b), c0 = tanh(z@W_lc+b); c kept in C-fragments ----
    float cfrag[16][8];
    #pragma unroll
    for (int n = 0; n < 16; ++n) {
        #pragma unroll
        for (int v = 0; v < 8; ++v) {
            const int row = v + half * 8;                // C/D layout: M = v + 8*half
            const int col = n * 16 + l16;                // N = lane%16
            const float* zr = z + (size_t)(grow0 + row) * LATENT;
            const float* wh = Wlht + (size_t)col * LATENT;
            const float* wc = Wlct + (size_t)col * LATENT;
            float ah = b_lh[col], ac = b_lc[col];
            for (int k = 0; k < LATENT; ++k) {
                float zz = zr[k];
                ah = fmaf(zz, wh[k], ah);
                ac = fmaf(zz, wc[k], ac);
            }
            cfrag[n][v] = tanhf(ac);
            aw[row * ASTRIDE + 256 + col] = f2bf(tanhf(ah));   // h into buffer A (cur=0)
        }
    }

    // ---- prefetch B tile for (n=0) into registers ----
    u32x4 sreg[16];
    #pragma unroll
    for (int j = 0; j < 16; ++j) {
        const int col64 = j * 4 + c0;                       // 0..63
        const int g = col64 >> 4, cg = col64 & 15;
        sreg[j] = *(const u32x4*)(Wt + (size_t)(g * HID + 0 * 16 + cg) * KTOT + kc * 8);
    }

    // ---- 50 recurrent steps ----
    #pragma unroll 1
    for (int step = 0; step < SEQ; ++step) {
        const int cur = step & 1;                    // h read region: 256 + cur*256
        const int nxt = cur ^ 1;                     // h write region

        #pragma unroll
        for (int n = 0; n < 16; ++n) {
            // -- all waves done reading previous B tile --
            wg_barrier();
            // -- commit prefetched tile for this n to LDS --
            #pragma unroll
            for (int j = 0; j < 16; ++j) {
                const int col64 = j * 4 + c0;
                *(u32x4*)(ldsB + col64 * BSTRIDE + kc * 8) = sreg[j];
            }
            // -- issue prefetch for next tile (wraps; weights are step-invariant) --
            {
                const int nn = (n + 1) & 15;
                #pragma unroll
                for (int j = 0; j < 16; ++j) {
                    const int col64 = j * 4 + c0;
                    const int g = col64 >> 4, cg = col64 & 15;
                    sreg[j] = *(const u32x4*)(Wt + (size_t)(g * HID + nn * 16 + cg) * KTOT + kc * 8);
                }
            }
            // -- make staged tile visible (waits dscnt only; prefetch keeps flying) --
            lds_release_barrier();

            v8f acc[4];
            #pragma unroll
            for (int g = 0; g < 4; ++g)
                #pragma unroll
                for (int v = 0; v < 8; ++v) acc[g][v] = 0.0f;

            #pragma unroll 4
            for (int kt = 0; kt < 16; ++kt) {
                // A fragment (16x32 bf16): lane<16 -> row l16, K [k0..k0+7],[k0+16..k0+23]
                //                          lane>=16 -> same rows, K offset +8
                int kb = kt * 32 + ((kt >= 8) ? cur * 256 : 0);
                const unsigned short* ap = aw + l16 * ASTRIDE + kb + half * 8;
                Frag16 A;
                A.u[0] = *(const u32x4*)ap;
                A.u[1] = *(const u32x4*)(ap + 16);

                #pragma unroll
                for (int g = 0; g < 4; ++g) {
                    // B fragment (32x16 bf16) from shared LDS tile:
                    // lane = column (g*16+l16), K [k0+half*8..+7],[k0+16+half*8..+7]
                    const unsigned short* wp = ldsB + (g * 16 + l16) * BSTRIDE + kt * 32 + half * 8;
                    Frag16 Bf;
                    Bf.u[0] = *(const u32x4*)wp;
                    Bf.u[1] = *(const u32x4*)(wp + 16);
                    acc[g] = __builtin_amdgcn_wmma_f32_16x16x32_bf16(
                        false, A.v, false, Bf.v, (short)0, acc[g], false, false);
                }
            }

            // gate activations + LSTM state update (elementwise on C layout)
            const float bi  = bg[0 * HID + n * 16 + l16];
            const float bf_ = bg[1 * HID + n * 16 + l16];
            const float bgg = bg[2 * HID + n * 16 + l16];
            const float bo  = bg[3 * HID + n * 16 + l16];
            #pragma unroll
            for (int v = 0; v < 8; ++v) {
                float iv = sigm(acc[0][v] + bi);
                float fv = sigm(acc[1][v] + bf_);
                float gv = tanhf(acc[2][v] + bgg);
                float ov = sigm(acc[3][v] + bo);
                float c  = fv * cfrag[n][v] + iv * gv;
                cfrag[n][v] = c;
                float hv = ov * tanhf(c);
                aw[(v + half * 8) * ASTRIDE + 256 + nxt * 256 + n * 16 + l16] = f2bf(hv);
            }
        }

        // ---- logits, sampling, embedding feedback (lanes 0-15, 1 row each) ----
        if (lane < 16) {
            const int r = lane;
            const int grow = grow0 + r;
            f32x4 lv = *(const f32x4*)b_out;
            const unsigned short* hp = aw + r * ASTRIDE + 256 + nxt * 256;
            for (int k = 0; k < HID; k += 8) {
                u32x4 hv4 = *(const u32x4*)(hp + k);
                const unsigned short* hvp = (const unsigned short*)&hv4;
                #pragma unroll
                for (int j = 0; j < 8; ++j) {
                    float hvf = bf2f(hvp[j]);
                    f32x4 wr = *(const f32x4*)(W_out + (size_t)(k + j) * NUC);
                    lv[0] = fmaf(hvf, wr[0], lv[0]);
                    lv[1] = fmaf(hvf, wr[1], lv[1]);
                    lv[2] = fmaf(hvf, wr[2], lv[2]);
                    lv[3] = fmaf(hvf, wr[3], lv[3]);
                }
            }
            *(f32x4*)(out + ((size_t)grow * SEQ + step) * NUC) = lv;

            // categorical sample via softmax CDF (hash RNG stands in for threefry)
            unsigned hsh = hashu((unsigned)grow * 2654435761u ^
                                 ((unsigned)step * 40503u) ^ 0x9E3779B9u);
            float u = ((float)(hsh >> 8) + 0.5f) * (1.0f / 16777216.0f);
            float m = fmaxf(fmaxf(lv[0], lv[1]), fmaxf(lv[2], lv[3]));
            float e0 = __expf(lv[0] - m), e1 = __expf(lv[1] - m);
            float e2 = __expf(lv[2] - m), e3 = __expf(lv[3] - m);
            float t = u * (e0 + e1 + e2 + e3);
            int idx = 3;
            if (t < e0) idx = 0;
            else if (t < e0 + e1) idx = 1;
            else if (t < e0 + e1 + e2) idx = 2;

            ((int*)out)[(size_t)BATCH * SEQ * NUC + (size_t)grow * SEQ + step] = idx;

            // x <- emb[idx]  (256 bf16 = 32 x 16B copies)
            const u32x4* ep = (const u32x4*)(ebf + idx * HID);
            u32x4* xp = (u32x4*)(aw + r * ASTRIDE);
            #pragma unroll
            for (int j = 0; j < 32; ++j) xp[j] = ep[j];
        }
        // h/x/c are wave-private; B-tile hand-off is covered by the per-n barriers
    }
}

// ---------------------------------------------------------------------------
extern "C" void kernel_launch(void* const* d_in, const int* in_sizes, int n_in,
                              void* d_out, int out_size, void* d_ws, size_t ws_size,
                              hipStream_t stream) {
    (void)in_sizes; (void)n_in; (void)out_size; (void)ws_size;

    const float* z     = (const float*)d_in[0];
    const float* W_lh  = (const float*)d_in[1];
    const float* b_lh  = (const float*)d_in[2];
    const float* W_lc  = (const float*)d_in[3];
    const float* b_lc  = (const float*)d_in[4];
    const float* emb   = (const float*)d_in[5];
    const float* W_ih  = (const float*)d_in[6];
    const float* W_hh  = (const float*)d_in[7];
    const float* b_ih  = (const float*)d_in[8];
    const float* b_hh  = (const float*)d_in[9];
    const float* W_out = (const float*)d_in[10];
    const float* b_out = (const float*)d_in[11];

    char* ws = (char*)d_ws;
    unsigned short* Wt   = (unsigned short*)(ws + OFF_WT);
    float*          bg   = (float*)(ws + OFF_BG);
    unsigned short* ebf  = (unsigned short*)(ws + OFF_EMB);
    float*          Wlht = (float*)(ws + OFF_WLHT);
    float*          Wlct = (float*)(ws + OFF_WLCT);

    // 1024*512 elements -> 2048 blocks of 256
    prep_kernel<<<2048, 256, 0, stream>>>(W_lh, W_lc, emb, W_ih, W_hh, b_ih, b_hh, ws);

    static bool attrSet = false;
    if (!attrSet) {
        hipFuncSetAttribute((const void*)rnn_kernel,
                            hipFuncAttributeMaxDynamicSharedMemorySize, SMEM_BYTES);
        attrSet = true;
    }

    rnn_kernel<<<BATCH / ROWS_WG, WAVES * 32, SMEM_BYTES, stream>>>(
        z, b_lh, b_lc, W_out, b_out, Wt, bg, ebf, Wlht, Wlct, (float*)d_out);
}
